// SLModel_84670985274130
// MI455X (gfx1250) — compile-verified, matrix operand order checked by reading, not run on hardware
//
#include <hip/hip_runtime.h>

// ---------------------------------------------------------------------------
// CDNA5 / gfx1250 LSTM seq2seq + BN-MLP heads.
// All matmuls: V_WMMA_F32_16X16X32_BF16 (wave32, f32 accumulate).
// All WMMA operands pre-converted to bf16 so fragments are two b128 loads.
// ---------------------------------------------------------------------------

#define TB    16384       // T*B rows (256 * 64)
#define TLEN  256
#define DIM   768
#define HID   512
#define GW    2048        // 4*H gate width
#define EMB   64
#define MHID  1024
#define HSTR  520         // LDS row stride (bf16) for h state: 1040B rows, 16B-aligned,
                          // 4-bank shift per row -> conflict-free b128 reads

typedef __attribute__((ext_vector_type(16))) __bf16 v16bf;
typedef __attribute__((ext_vector_type(8)))  __bf16 v8bf;
typedef __attribute__((ext_vector_type(8)))  float  v8f;

#define CAT16(lo, hi) __builtin_shufflevector((lo), (hi), 0,1,2,3,4,5,6,7,8,9,10,11,12,13,14,15)

__device__ __forceinline__ __bf16 f2bf(float f) {
  unsigned int u = __builtin_bit_cast(unsigned int, f);
  unsigned int r = (u + 0x7FFFu + ((u >> 16) & 1u)) >> 16;
  unsigned short s = (unsigned short)r;
  return __builtin_bit_cast(__bf16, s);
}

__device__ __forceinline__ float sigm(float x) { return 1.0f / (1.0f + expf(-x)); }

// ---------------------------------------------------------------------------
// bf16 WMMA GEMM:  C[r, n] (+=) sum_k A(r,k)*B(n,k)  + bias0[n] + bias1[n]
// A: bf16, rows r in [0,TB), A(r,k) = A[(r/64)*aS0 + (r%64)*aS1 + k] (k contig).
// B: bf16 [N, K] row-major with leading dim bLd.
// One wave per 64x16 output strip (4 M-tiles share each B fragment).
// N must be a multiple of 16 (guaranteed by weight padding). No divergence.
// ---------------------------------------------------------------------------
__global__ void k_gemm_bf(const __bf16* __restrict__ A, long aS0, long aS1,
                          const __bf16* __restrict__ B, long bLd,
                          const float* __restrict__ bias0, const float* __restrict__ bias1,
                          float* __restrict__ C, int ldc, int K, int addC) {
  const int lane = threadIdx.x & 31;
  const int half = lane >> 4;
  const int lr   = lane & 15;
  const int m0   = blockIdx.y * 64;
  const int nb   = blockIdx.x * 16 + lr;
  const int koff = 8 * half;

  long abase[4];
#pragma unroll
  for (int i = 0; i < 4; ++i) {
    int ra = m0 + i * 16 + lr;
    abase[i] = (long)(ra >> 6) * aS0 + (long)(ra & 63) * aS1;
  }
  const __bf16* __restrict__ brow = B + (long)nb * bLd;

  v8f acc[4];
#pragma unroll
  for (int i = 0; i < 4; ++i)
#pragma unroll
    for (int r = 0; r < 8; ++r)
      acc[i][r] = addC ? C[(long)(m0 + i * 16 + r + 8 * half) * ldc + nb] : 0.0f;

  for (int k0 = 0; k0 < K; k0 += 32) {
    __builtin_prefetch(brow + k0 + 256, 0, 3);
    v8bf blo = *(const v8bf*)(brow + k0 + koff);
    v8bf bhi = *(const v8bf*)(brow + k0 + 16 + koff);
    v16bf b = CAT16(blo, bhi);
#pragma unroll
    for (int i = 0; i < 4; ++i) {
      v8bf alo = *(const v8bf*)(A + abase[i] + k0 + koff);
      v8bf ahi = *(const v8bf*)(A + abase[i] + k0 + 16 + koff);
      v16bf a = CAT16(alo, ahi);
      acc[i] = __builtin_amdgcn_wmma_f32_16x16x32_bf16(false, a, false, b,
                                                       (short)0, acc[i], false, false);
    }
  }

  float badd = 0.0f;
  if (bias0) badd += bias0[nb];
  if (bias1) badd += bias1[nb];
#pragma unroll
  for (int i = 0; i < 4; ++i)
#pragma unroll
    for (int r = 0; r < 8; ++r)
      C[(long)(m0 + i * 16 + r + 8 * half) * ldc + nb] = acc[i][r] + badd;
}

// ---------------------------------------------------------------------------
// Persistent LSTM recurrence. One 32-wave workgroup per direction.
// G (fp32) holds precomputed input gates incl. biases. W is bf16 [2048, 512].
// h state: bf16 in LDS [64][HSTR]; c state: fp32 registers in D-frag layout.
// gridDim.x = 1 (unidir) or 2 (bidir; block 1 = reverse, writes cols 512..1023).
// ---------------------------------------------------------------------------
__global__ void k_lstm(const float* __restrict__ Gf, const float* __restrict__ Gb,
                       const __bf16* __restrict__ Wf, const __bf16* __restrict__ Wb,
                       __bf16* __restrict__ out, int out_ld) {
  extern __shared__ __bf16 sh[];
  const int dir = blockIdx.x;
  const float*  __restrict__ G = dir ? Gb : Gf;
  const __bf16* __restrict__ W = dir ? Wb : Wf;
  const int colOff = dir * HID;

  const int tid  = threadIdx.x;
  const int lane = tid & 31;
  const int w    = tid >> 5;
  const int half = lane >> 4;
  const int lr   = lane & 15;
  const int koff = 8 * half;

  for (int i = tid; i < 64 * HSTR; i += 1024) sh[i] = f2bf(0.0f);
  __syncthreads();

  const int gi0  = w * 4;                       // 4 (mt,jt) tile groups per wave
  const int mt   = gi0 >> 5;                    // same m-tile for the whole wave
  const __bf16* __restrict__ arow = sh + (mt * 16 + lr) * HSTR;

  float cst[4][8];
#pragma unroll
  for (int g = 0; g < 4; ++g)
#pragma unroll
    for (int r = 0; r < 8; ++r) cst[g][r] = 0.0f;

  for (int s = 0; s < TLEN; ++s) {
    const int t = dir ? (TLEN - 1 - s) : s;
    float hst[4][8];

#pragma unroll
    for (int g = 0; g < 4; ++g) {
      const int jc = ((gi0 + g) & 31) * 16 + lr;     // cell column 0..511

      v8f acc[4];                                    // i, f, g, o
#pragma unroll
      for (int r = 0; r < 8; ++r) {
        long gb = (long)(t * 64 + mt * 16 + r + 8 * half) * GW;
#pragma unroll
        for (int q = 0; q < 4; ++q) acc[q][r] = G[gb + q * HID + jc];
      }

      for (int k0 = 0; k0 < HID; k0 += 32) {
        v8bf alo = *(const v8bf*)(arow + k0 + koff);
        v8bf ahi = *(const v8bf*)(arow + k0 + 16 + koff);
        v16bf a = CAT16(alo, ahi);
#pragma unroll
        for (int q = 0; q < 4; ++q) {
          const __bf16* __restrict__ wr = W + (long)(q * HID + jc) * HID;
          v8bf blo = *(const v8bf*)(wr + k0 + koff);
          v8bf bhi = *(const v8bf*)(wr + k0 + 16 + koff);
          v16bf b = CAT16(blo, bhi);
          acc[q] = __builtin_amdgcn_wmma_f32_16x16x32_bf16(false, a, false, b,
                                                           (short)0, acc[q], false, false);
        }
      }

#pragma unroll
      for (int r = 0; r < 8; ++r) {
        float iv = sigm(acc[0][r]);
        float fv = sigm(acc[1][r]);
        float gv = tanhf(acc[2][r]);
        float ov = sigm(acc[3][r]);
        float cc = fv * cst[g][r] + iv * gv;
        cst[g][r] = cc;
        hst[g][r] = ov * tanhf(cc);
      }
    }

    __syncthreads();                                 // all reads of old h done
#pragma unroll
    for (int g = 0; g < 4; ++g) {
      const int jc = ((gi0 + g) & 31) * 16 + lr;
#pragma unroll
      for (int r = 0; r < 8; ++r) {
        int m = mt * 16 + r + 8 * half;
        __bf16 hb = f2bf(hst[g][r]);
        sh[m * HSTR + jc] = hb;
        out[(long)(t * 64 + m) * out_ld + colOff + jc] = hb;
      }
    }
    __syncthreads();                                 // new h visible
  }
}

// ---------------------------------------------------------------------------
// fp32 -> bf16 elementwise convert.
__global__ void k_cvt(const float* __restrict__ src, __bf16* __restrict__ dst, int n) {
  int i = blockIdx.x * 256 + threadIdx.x;
  if (i < n) dst[i] = f2bf(src[i]);
}

// Transpose fp32 [Kd, Nd] -> bf16 [Nd, Kd].
__global__ void k_tr(const float* __restrict__ src, __bf16* __restrict__ dst,
                     int Kd, int Nd) {
  int i = blockIdx.x * 256 + threadIdx.x;
  if (i >= Kd * Nd) return;
  int n = i / Kd, k = i % Kd;
  dst[i] = f2bf(src[(long)k * Nd + n]);
}

// Transpose + pad fp32 [Kd, srcN] -> bf16 [16, Kd] (cols >= srcN are zero).
__global__ void k_trp(const float* __restrict__ src, __bf16* __restrict__ dst,
                      int Kd, int srcN) {
  int i = blockIdx.x * 256 + threadIdx.x;
  if (i >= 16 * Kd) return;
  int n = i / Kd, k = i % Kd;
  dst[i] = (n < srcN) ? f2bf(src[(long)k * srcN + n]) : f2bf(0.0f);
}

// Teacher-forced tag embedding (bf16 out): te[t*64+b,:] = emb[t==0 ? 8 : lbl[b,t-1]]
__global__ void k_embed(const int* __restrict__ labels, const float* __restrict__ emb,
                        __bf16* __restrict__ te) {
  int i = blockIdx.x * 256 + threadIdx.x;            // TB*EMB exact
  int r = i >> 6, e = i & 63;
  int t = r >> 6, b = r & 63;
  int id = (t == 0) ? 8 : labels[b * TLEN + (t - 1)];
  te[i] = f2bf(emb[id * EMB + e]);
}

// BatchNorm training-mode batch stats over TB rows (biased var).
__global__ void k_bnstats(const float* __restrict__ Hb, float* __restrict__ mu,
                          float* __restrict__ var) {
  __shared__ float s1[256], s2[256];
  int c = blockIdx.x, tid = threadIdx.x;
  float a = 0.0f, b = 0.0f;
  for (int r = tid; r < TB; r += 256) {
    float v = Hb[(long)r * MHID + c];
    a += v; b += v * v;
  }
  s1[tid] = a; s2[tid] = b;
  __syncthreads();
  for (int st = 128; st > 0; st >>= 1) {
    if (tid < st) { s1[tid] += s1[tid + st]; s2[tid] += s2[tid + st]; }
    __syncthreads();
  }
  if (tid == 0) {
    float m = s1[0] / (float)TB;
    mu[c] = m;
    var[c] = s2[0] / (float)TB - m * m;
  }
}

// Normalize + LeakyReLU; emit bf16 activations for the second head GEMM.
__global__ void k_bnapply(const float* __restrict__ Hb, __bf16* __restrict__ outbf,
                          const float* __restrict__ mu, const float* __restrict__ var,
                          const float* __restrict__ g, const float* __restrict__ be) {
  long i = (long)blockIdx.x * 256 + threadIdx.x;
  if (i >= (long)TB * MHID) return;
  int c = (int)(i & (MHID - 1));
  float h = Hb[i];
  h = g[c] * (h - mu[c]) * rsqrtf(var[c] + 1e-5f) + be[c];
  outbf[i] = f2bf(h > 0.0f ? h : 0.01f * h);
}

// log_softmax over `width` valid logits (ld 16); bias b2 folded in here.
__global__ void k_lsm(const float* __restrict__ logits, const float* __restrict__ b2,
                      int width, float* __restrict__ outp) {
  int r = blockIdx.x * blockDim.x + threadIdx.x;
  if (r >= TB) return;
  float v[8];
  float mx = -3.4e38f;
  for (int j = 0; j < width; ++j) { v[j] = logits[r * 16 + j] + b2[j]; mx = fmaxf(mx, v[j]); }
  float s = 0.0f;
  for (int j = 0; j < width; ++j) s += expf(v[j] - mx);
  float lse = mx + logf(s);
  for (int j = 0; j < width; ++j) outp[(long)r * width + j] = v[j] - lse;
}

// ---------------------------------------------------------------------------
extern "C" void kernel_launch(void* const* d_in, const int* in_sizes, int n_in,
                              void* d_out, int out_size, void* d_ws, size_t ws_size,
                              hipStream_t stream) {
  (void)in_sizes; (void)n_in; (void)out_size; (void)ws_size;

  const float* x          = (const float*)d_in[0];
  const int*   tag_labels = (const int*)  d_in[1];
  const float* e0Wih = (const float*)d_in[2];  const float* e0Whh = (const float*)d_in[3];
  const float* e0bih = (const float*)d_in[4];  const float* e0bhh = (const float*)d_in[5];
  const float* e1Wih = (const float*)d_in[6];  const float* e1Whh = (const float*)d_in[7];
  const float* e1bih = (const float*)d_in[8];  const float* e1bhh = (const float*)d_in[9];
  const float* d0Wih = (const float*)d_in[10]; const float* d0Whh = (const float*)d_in[11];
  const float* d0bih = (const float*)d_in[12]; const float* d0bhh = (const float*)d_in[13];
  const float* d1Wih = (const float*)d_in[14]; const float* d1Whh = (const float*)d_in[15];
  const float* d1bih = (const float*)d_in[16]; const float* d1bhh = (const float*)d_in[17];
  const float* temb  = (const float*)d_in[18];

  // ---- workspace: fp32 region (gate buffers, reused for head scratch) ------
  float* ws = (float*)d_ws;
  float* G0 = ws;                                   // [TB, 2048]
  float* G1 = G0 + (size_t)TB * GW;                 // [TB, 2048]
  float* HB  = G0;                                  // head hidden fp32 [TB, 1024]
  float* MU  = G1;                                  // [1024]
  float* VAR = G1 + 1024;                           // [1024]
  float* LOG = G1 + 2048;                           // [TB, 16]

  // ---- workspace: bf16 region ----------------------------------------------
  __bf16* bp = (__bf16*)(G1 + (size_t)TB * GW);
  auto abf = [&](size_t n) { __bf16* p = bp; bp += n; return p; };
  __bf16* Xbf   = abf((size_t)64 * TLEN * DIM);
  __bf16* E0Wih = abf((size_t)2 * GW * DIM);
  __bf16* E0Whh = abf((size_t)2 * GW * HID);
  __bf16* E1Wih = abf((size_t)2 * GW * MHID);
  __bf16* E1Whh = abf((size_t)2 * GW * HID);
  __bf16* D0Wih = abf((size_t)GW * 1088);
  __bf16* D0Whh = abf((size_t)GW * HID);
  __bf16* D1Wih = abf((size_t)GW * HID);
  __bf16* D1Whh = abf((size_t)GW * HID);
  __bf16* W1T[3] = { abf((size_t)MHID * HID), abf((size_t)MHID * HID), abf((size_t)MHID * HID) };
  __bf16* W2T[3] = { abf((size_t)16 * MHID), abf((size_t)16 * MHID), abf((size_t)16 * MHID) };
  __bf16* TEbf  = abf((size_t)TB * EMB);
  __bf16* H0bf  = abf((size_t)TB * MHID);
  __bf16* H1bf  = abf((size_t)TB * MHID);
  __bf16* HBbf  = abf((size_t)TB * MHID);

  const size_t SHBYTES = (size_t)64 * HSTR * sizeof(unsigned short);

  auto cvt = [stream](const float* s, __bf16* d, int n) {
    k_cvt<<<(n + 255) / 256, 256, 0, stream>>>(s, d, n);
  };
  auto gemm = [stream](const __bf16* A, long aS0, long aS1, const __bf16* B, long bLd,
                       const float* b0, const float* b1, float* C, int ldc,
                       int Nn, int K, int addC) {
    dim3 grid(Nn / 16, TB / 64);
    k_gemm_bf<<<grid, 32, 0, stream>>>(A, aS0, aS1, B, bLd, b0, b1, C, ldc, K, addC);
  };

  // ---- one-time (per launch) operand conversion ----------------------------
  cvt(x, Xbf, 64 * TLEN * DIM);
  cvt(e0Wih, E0Wih, 2 * GW * DIM);   cvt(e0Whh, E0Whh, 2 * GW * HID);
  cvt(e1Wih, E1Wih, 2 * GW * MHID);  cvt(e1Whh, E1Whh, 2 * GW * HID);
  cvt(d0Wih, D0Wih, GW * 1088);      cvt(d0Whh, D0Whh, GW * HID);
  cvt(d1Wih, D1Wih, GW * HID);       cvt(d1Whh, D1Whh, GW * HID);
  for (int h = 0; h < 3; ++h) {
    const float* W1 = (const float*)d_in[19 + 6 * h];
    const float* W2 = (const float*)d_in[23 + 6 * h];
    int width = (h == 0) ? 8 : 2;
    k_tr <<<(HID * MHID + 255) / 256, 256, 0, stream>>>(W1, W1T[h], HID, MHID);
    k_trp<<<(16 * MHID + 255) / 256, 256, 0, stream>>>(W2, W2T[h], MHID, width);
  }
  k_embed<<<(TB * EMB) / 256, 256, 0, stream>>>(tag_labels, temb, TEbf);

  // ---- encoder layer 0 (bidir); x[b,t,:] -> row r = t*64+b -----------------
  gemm(Xbf, (long)DIM, (long)TLEN * DIM, E0Wih,            DIM, e0bih,      e0bhh,      G0, GW, GW, DIM, 0);
  gemm(Xbf, (long)DIM, (long)TLEN * DIM, E0Wih + GW * DIM, DIM, e0bih + GW, e0bhh + GW, G1, GW, GW, DIM, 0);
  k_lstm<<<2, 1024, SHBYTES, stream>>>(G0, G1, E0Whh, E0Whh + GW * HID, H0bf, 1024);

  // ---- encoder layer 1 (bidir) ---------------------------------------------
  gemm(H0bf, 64L * 1024, 1024, E1Wih,             MHID, e1bih,      e1bhh,      G0, GW, GW, MHID, 0);
  gemm(H0bf, 64L * 1024, 1024, E1Wih + GW * MHID, MHID, e1bih + GW, e1bhh + GW, G1, GW, GW, MHID, 0);
  k_lstm<<<2, 1024, SHBYTES, stream>>>(G0, G1, E1Whh, E1Whh + GW * HID, H1bf, 1024);

  // ---- decoder layer 0: concat(H1, TE) via two fused GEMMs -----------------
  gemm(H1bf, 64L * 1024, 1024, D0Wih,        1088, d0bih, d0bhh, G0, GW, GW, MHID, 0);
  gemm(TEbf, 64L * EMB,  EMB,  D0Wih + 1024, 1088, nullptr, nullptr, G0, GW, GW, EMB, 1);
  k_lstm<<<1, 1024, SHBYTES, stream>>>(G0, nullptr, D0Whh, nullptr, H0bf, HID);

  // ---- decoder layer 1 ------------------------------------------------------
  gemm(H0bf, 64L * HID, HID, D1Wih, HID, d1bih, d1bhh, G1, GW, GW, HID, 0);
  k_lstm<<<1, 1024, SHBYTES, stream>>>(G1, nullptr, D1Whh, nullptr, H1bf, HID);

  // ---- heads ----------------------------------------------------------------
  long ooff[3] = { 0, (long)TB * 8, (long)TB * 10 };
  for (int h = 0; h < 3; ++h) {
    const float* b1 = (const float*)d_in[20 + 6 * h];
    const float* g  = (const float*)d_in[21 + 6 * h];
    const float* be = (const float*)d_in[22 + 6 * h];
    const float* b2 = (const float*)d_in[24 + 6 * h];
    int width = (h == 0) ? 8 : 2;
    gemm(H1bf, 64L * HID, HID, W1T[h], HID, b1, nullptr, HB, MHID, MHID, HID, 0);
    k_bnstats<<<MHID, 256, 0, stream>>>(HB, MU, VAR);
    k_bnapply<<<(TB * MHID) / 256, 256, 0, stream>>>(HB, HBbf, MU, VAR, g, be);
    gemm(HBbf, 64L * MHID, MHID, W2T[h], MHID, nullptr, nullptr, LOG, 16, 16, MHID, 0);
    k_lsm<<<TB / 256, 256, 0, stream>>>(LOG, b2, width, (float*)d_out + ooff[h]);
  }
}
